// VisionTransformerSingle_23244363005957
// MI455X (gfx1250) — matrix operand order repaired and checked by reading, not compiled, not run on hardware
//
#include <hip/hip_runtime.h>
#include <hip/hip_bf16.h>

// ---------------- types ----------------
typedef _Float16 f16;
typedef __attribute__((ext_vector_type(16))) _Float16 v16h;
typedef __attribute__((ext_vector_type(8)))  _Float16 v8h;
typedef __attribute__((ext_vector_type(8)))  float    v8f;
typedef __attribute__((ext_vector_type(4)))  int      v4i;

// CDNA5 async global->LDS path (ASYNCcnt), with sync fallback
#if defined(__has_builtin)
# if __has_builtin(__builtin_amdgcn_global_load_async_to_lds_b128) && \
     __has_builtin(__builtin_amdgcn_s_wait_asynccnt)
#  define USE_ASYNC_LDS 1
# endif
#endif
#ifndef USE_ASYNC_LDS
# define USE_ASYNC_LDS 0
#endif

#define AS_GLOBAL __attribute__((address_space(1)))
#define AS_LDS    __attribute__((address_space(3)))

// ---------------- model dims ----------------
#define BB      128
#define CC      3
#define IMG     224
#define NPATCH  196          // (224/16)^2
#define SS      197          // 1 + NPATCH
#define DD      384
#define HEADS   6
#define HDIM    64
#define DEPTH   12
#define HID     1536
#define NCLS    100
#define PD      768          // C*P*P
#define M_TOK   (BB*SS)      // 25216
#define M_PATCH (BB*NPATCH)  // 25088
#define SPAD    224          // S padded to multiple of 32

#define LDP 40               // padded LDS row pitch (f16), keeps 16B alignment
#define BM  128              // block tile M
#define BN  64               // block tile N

enum { EPI_F32 = 0, EPI_F16 = 1, EPI_F16_GELU = 2 };

// =====================================================================
// Tiled WMMA GEMM:  C = A(f16,[M,K]) * B^T(or B) + bias (+resid)
//   BT=false: B is f16 [N, ldb], contraction over its 2nd axis (weight form)
//   BT=true : B is f16 [K, ldb], contraction over its 1st axis (P @ V form)
// 128x64 tile / block, 8 waves, each wave 32x32 (2x2 WMMA tiles, 4 wmma per
// BK=32 step), double-buffered LDS (1 barrier per k-step), branchless
// clamped staging. On gfx1250 toolchains with the async builtins, staging
// uses global_load_async_to_lds_b128 (ASYNCcnt) instead of the VGPR bounce.
// =====================================================================
template<int EPI, bool BT>
__global__ __launch_bounds__(256) void gemm_wmma(
    const f16* __restrict__ A, long long aBatch, int lda,
    const f16* __restrict__ B, long long bBatch, int ldb,
    const float* __restrict__ bias,
    const float* __restrict__ resid,
    void* __restrict__ Cv,
    long long cOuter, long long cInner, int innerCnt, int ldc, int colOff,
    int M, int N, int K, int Kact,
    int rmapDiv, int rmapStride, int rmapOff)
{
    __shared__ f16 As[2][BM * LDP];
    __shared__ f16 Bs[2][BN * LDP];

    const int tid  = threadIdx.x;
    const int lane = tid & 31;
    const int wave = tid >> 5;   // 0..7
    const int wm   = wave & 3;   // 4 row groups of 32
    const int wn   = wave >> 2;  // 2 col groups of 32
    const int z    = blockIdx.z;
    const int m0   = blockIdx.y * BM;
    const int n0   = blockIdx.x * BN;

    const f16* Ab = A + (long long)z * aBatch;
    const f16* Bb = B + (long long)z * bBatch;
    const long long cbase = (innerCnt > 0)
        ? (long long)(z / innerCnt) * cOuter + (long long)(z % innerCnt) * cInner
        : (long long)z * cOuter;

    v8f acc00 = {}, acc01 = {}, acc10 = {}, acc11 = {};

    const int half16 = lane >> 4;    // 0 | 1
    const int lrow   = lane & 15;

    // one b128: global (clamped, full-EXEC) -> LDS
    auto copy128 = [&](const f16* src, f16* dst) {
#if USE_ASYNC_LDS
        __builtin_amdgcn_global_load_async_to_lds_b128(
            (AS_GLOBAL v4i*)(AS_GLOBAL void*)(void*)const_cast<f16*>(src),
            (AS_LDS    v4i*)(AS_LDS    void*)(void*)dst, 0, 0);
#else
        v8h v = *(const v8h*)src;
        *(v8h*)dst = v;
#endif
    };

    // ---- staging helpers (branchless clamped loads) ----
    auto stageA = [&](int k0, int buf) {
#pragma unroll
        for (int p = 0; p < 2; ++p) {
            const int row = p * 64 + (tid >> 2);   // 0..127
            const int col = (tid & 3) * 8;         // 0,8,16,24
            const int gm  = m0 + row;
            const int gmc = gm < M ? gm : M - 1;   // clamp: junk rows -> discarded C rows
            const f16* src = Ab + (long long)gmc * lda + k0 + col;
            __builtin_prefetch(src + 64, 0, 3);    // speculative, next-next K tile
            copy128(src, &As[buf][row * LDP + col]);
        }
    };
    auto stageB = [&](int k0, int buf) {
        if (!BT) {
            const int row = tid >> 2;              // 0..63 (= n within tile)
            const int col = (tid & 3) * 8;
            const int gn  = n0 + row;
            const int gnc = gn < N ? gn : N - 1;   // clamp: junk rows -> discarded C cols
            copy128(Bb + (long long)gnc * ldb + k0 + col, &Bs[buf][row * LDP + col]);
        } else {
            const int kk = tid >> 3;               // 0..31 (= k within tile)
            const int ng = (tid & 7) * 8;          // 0..56
            const int gk = k0 + kk;
            v8h bv = {};
            if (gk < Kact)                         // contraction dim: must be exact zero
                bv = *(const v8h*)(Bb + (long long)gk * ldb + n0 + ng);
#pragma unroll
            for (int i = 0; i < 8; ++i)
                Bs[buf][(ng + i) * LDP + kk] = bv[i];
        }
    };
    auto compute = [&](int buf) {
        v16h af[2], bf[2];
#pragma unroll
        for (int tm = 0; tm < 2; ++tm) {
            // A frag: lanes<16 K={0..7,16..23}, lanes>=16 K={8..15,24..31}
            const f16* ap = &As[buf][(wm * 32 + tm * 16 + lrow) * LDP + half16 * 8];
            v8h lo = *(const v8h*)(ap);
            v8h hi = *(const v8h*)(ap + 16);
#pragma unroll
            for (int i = 0; i < 8; ++i) { af[tm][i] = lo[i]; af[tm][8 + i] = hi[i]; }
        }
#pragma unroll
        for (int tn = 0; tn < 2; ++tn) {
            // B frag: lanes<16 K={0..7,16..23}, lanes>=16 K={8..15,24..31}
            const f16* bp = &Bs[buf][(wn * 32 + tn * 16 + lrow) * LDP + half16 * 8];
            v8h lo = *(const v8h*)(bp);
            v8h hi = *(const v8h*)(bp + 16);
#pragma unroll
            for (int i = 0; i < 8; ++i) { bf[tn][i] = lo[i]; bf[tn][8 + i] = hi[i]; }
        }
        acc00 = __builtin_amdgcn_wmma_f32_16x16x32_f16(false, af[0], false, bf[0], (short)0, acc00, false, false);
        acc01 = __builtin_amdgcn_wmma_f32_16x16x32_f16(false, af[0], false, bf[1], (short)0, acc01, false, false);
        acc10 = __builtin_amdgcn_wmma_f32_16x16x32_f16(false, af[1], false, bf[0], (short)0, acc10, false, false);
        acc11 = __builtin_amdgcn_wmma_f32_16x16x32_f16(false, af[1], false, bf[1], (short)0, acc11, false, false);
    };

    // ---- main loop: double buffered, one barrier per k-step ----
    stageA(0, 0);
    stageB(0, 0);
    int buf = 0;
    for (int k0 = 0; k0 < K; k0 += 32) {
#if USE_ASYNC_LDS
        __builtin_amdgcn_s_wait_asynccnt(0);   // own async fills done before barrier
#endif
        __syncthreads();                 // staged tile `buf` ready; reads of buf^1 done
        const int kn = k0 + 32;
        if (kn < K) { stageA(kn, buf ^ 1); stageB(kn, buf ^ 1); }
        compute(buf);
        buf ^= 1;
    }

    // ---- epilogue ----
#pragma unroll
    for (int tm = 0; tm < 2; ++tm) {
#pragma unroll
        for (int r = 0; r < 8; ++r) {
            const int gm = m0 + wm * 32 + tm * 16 + half16 * 8 + r;
            if (gm >= M) continue;
            long long orow = gm;
            if (rmapDiv > 0)
                orow = (long long)(gm / rmapDiv) * rmapStride + rmapOff + (gm % rmapDiv);
#pragma unroll
            for (int tn = 0; tn < 2; ++tn) {
                const int gn = n0 + wn * 32 + tn * 16 + lrow;
                if (gn >= N) continue;
                float v;
                if (tm == 0) v = (tn == 0) ? acc00[r] : acc01[r];
                else         v = (tn == 0) ? acc10[r] : acc11[r];
                if (bias) v += bias[gn];
                const long long cidx = cbase + orow * (long long)ldc + colOff + gn;
                if (EPI == EPI_F32) {
                    float* C = (float*)Cv;
                    if (resid) v += resid[cidx];
                    C[cidx] = v;
                } else if (EPI == EPI_F16) {
                    ((f16*)Cv)[cidx] = (f16)v;
                } else { // exact GELU
                    const float g = 0.5f * v * (1.0f + erff(v * 0.70710678118f));
                    ((f16*)Cv)[cidx] = (f16)g;
                }
            }
        }
    }
}

// =====================================================================
// LayerNorm over D=384: f32 in (arbitrary row stride) -> f16 out (ld 384)
// =====================================================================
__global__ __launch_bounds__(128) void ln_kernel(
    const float* __restrict__ X, long long rowStride,
    const float* __restrict__ g, const float* __restrict__ b,
    f16* __restrict__ Y)
{
    const long long row = blockIdx.x;
    const float* x = X + row * rowStride;
    f16* y = Y + row * (long long)DD;
    __shared__ float red[128];
    const int t = threadIdx.x;
    const float v0 = x[t], v1 = x[t + 128], v2 = x[t + 256];
    red[t] = v0 + v1 + v2;
    __syncthreads();
    for (int o = 64; o > 0; o >>= 1) { if (t < o) red[t] += red[t + o]; __syncthreads(); }
    const float mean = red[0] * (1.0f / 384.0f);
    __syncthreads();
    const float d0 = v0 - mean, d1 = v1 - mean, d2 = v2 - mean;
    red[t] = d0 * d0 + d1 * d1 + d2 * d2;
    __syncthreads();
    for (int o = 64; o > 0; o >>= 1) { if (t < o) red[t] += red[t + o]; __syncthreads(); }
    const float rstd = rsqrtf(red[0] * (1.0f / 384.0f) + 1e-5f);
    y[t]       = (f16)(d0 * rstd * g[t]       + b[t]);
    y[t + 128] = (f16)(d1 * rstd * g[t + 128] + b[t + 128]);
    y[t + 256] = (f16)(d2 * rstd * g[t + 256] + b[t + 256]);
}

// =====================================================================
// In-place softmax over one padded score row (f16, S valid, ld=SPAD)
// =====================================================================
__global__ __launch_bounds__(256) void softmax_kernel(f16* __restrict__ P)
{
    f16* p = P + (long long)blockIdx.x * SPAD;
    const int t = threadIdx.x;
    __shared__ float red[256];
    const float v = (t < SS) ? (float)p[t] : -3.0e38f;
    red[t] = v;
    __syncthreads();
    for (int o = 128; o > 0; o >>= 1) { if (t < o) red[t] = fmaxf(red[t], red[t + o]); __syncthreads(); }
    const float mx = red[0];
    __syncthreads();
    const float e = (t < SS) ? __expf(v - mx) : 0.0f;
    red[t] = e;
    __syncthreads();
    for (int o = 128; o > 0; o >>= 1) { if (t < o) red[t] += red[t + o]; __syncthreads(); }
    const float inv = 1.0f / red[0];
    if (t < SPAD) p[t] = (f16)(e * inv);   // pad cols get exact 0
}

// =====================================================================
// QKV split into per-head [B*H, S, 64] f16 buffers; scale folded into Q
// =====================================================================
__global__ void qkv_split_kernel(const f16* __restrict__ qkv,
                                 f16* __restrict__ Qh, f16* __restrict__ Kh,
                                 f16* __restrict__ Vh)
{
    const long long idx = (long long)blockIdx.x * 256 + threadIdx.x;
    if (idx >= (long long)M_TOK * (3 * DD)) return;
    const int d3 = (int)(idx % (3 * DD));
    const long long row = idx / (3 * DD);          // b*197 + s
    const int b = (int)(row / SS), s = (int)(row % SS);
    const int which = d3 / DD;
    const int d = d3 % DD;
    const int hh = d >> 6, dd = d & 63;
    const long long o = (((long long)(b * HEADS + hh) * SS) + s) * HDIM + dd;
    const f16 v = qkv[idx];
    if (which == 0)      Qh[o] = (f16)((float)v * 0.125f);  // 1/sqrt(64)
    else if (which == 1) Kh[o] = v;
    else                 Vh[o] = v;
}

// =====================================================================
// im2col patch gather: x[B,C,224,224] f32 -> Apatch[B*196, 768] f16
// =====================================================================
__global__ void im2col_kernel(const float* __restrict__ x, f16* __restrict__ Ap)
{
    const long long idx = (long long)blockIdx.x * 256 + threadIdx.x;
    if (idx >= (long long)M_PATCH * PD) return;
    const int k = (int)(idx % PD);
    const long long m = idx / PD;
    const int b = (int)(m / NPATCH), p = (int)(m % NPATCH);
    const int py = p / 14, px = p % 14;
    const int c = k / 256, r = k % 256;
    const int iy = r / 16, ix = r % 16;
    const long long src = (((long long)(b * CC + c) * IMG) + py * 16 + iy) * IMG + px * 16 + ix;
    Ap[idx] = (f16)x[src];
}

__global__ void f32_to_f16_kernel(const float* __restrict__ src, f16* __restrict__ dst, long long n)
{
    const long long i = (long long)blockIdx.x * 256 + threadIdx.x;
    if (i < n) dst[i] = (f16)src[i];
}

// h[b,0,:] = cls + pos[0];  h[b,1+p,:] += pos[1+p]
__global__ void pos_cls_kernel(float* __restrict__ hbuf,
                               const float* __restrict__ cls,
                               const float* __restrict__ pos)
{
    const long long idx = (long long)blockIdx.x * 256 + threadIdx.x;
    if (idx >= (long long)M_TOK * DD) return;
    const int d = (int)(idx % DD);
    const long long row = idx / DD;
    const int s = (int)(row % SS);
    if (s == 0) hbuf[idx] = cls[d] + pos[d];
    else        hbuf[idx] += pos[(long long)s * DD + d];
}

// ---------------- host side ----------------
static inline char* bump(char*& p, size_t bytes) {
    char* r = p;
    p += (bytes + 255) & ~(size_t)255;
    return r;
}
static inline int cdiv(long long a, long long b) { return (int)((a + b - 1) / b); }

extern "C" void kernel_launch(void* const* d_in, const int* in_sizes, int n_in,
                              void* d_out, int out_size, void* d_ws, size_t ws_size,
                              hipStream_t stream) {
    (void)in_sizes; (void)n_in; (void)out_size; (void)ws_size;
    const float* x       = (const float*)d_in[0];
    const float* patch_w = (const float*)d_in[1];
    const float* patch_b = (const float*)d_in[2];
    const float* cls_tok = (const float*)d_in[3];
    const float* pos_emb = (const float*)d_in[4];
    const float* ln1_g   = (const float*)d_in[5];
    const float* ln1_b   = (const float*)d_in[6];
    const float* qkv_w   = (const float*)d_in[7];
    const float* qkv_b   = (const float*)d_in[8];
    const float* out_w   = (const float*)d_in[9];
    const float* out_b   = (const float*)d_in[10];
    const float* ln2_g   = (const float*)d_in[11];
    const float* ln2_b   = (const float*)d_in[12];
    const float* fc1_w   = (const float*)d_in[13];
    const float* fc1_b   = (const float*)d_in[14];
    const float* fc2_w   = (const float*)d_in[15];
    const float* fc2_b   = (const float*)d_in[16];
    const float* lnf_g   = (const float*)d_in[17];
    const float* lnf_b   = (const float*)d_in[18];
    const float* head_w  = (const float*)d_in[19];
    float* outp = (float*)d_out;

    // ----- workspace layout -----
    char* wp = (char*)d_ws;
    f16*   apatch = (f16*)bump(wp, (size_t)M_PATCH * PD * 2);
    f16*   w_pat  = (f16*)bump(wp, (size_t)DD * PD * 2);
    f16*   w_qkv  = (f16*)bump(wp, (size_t)DEPTH * 3 * DD * DD * 2);
    f16*   w_out  = (f16*)bump(wp, (size_t)DEPTH * DD * DD * 2);
    f16*   w_fc1  = (f16*)bump(wp, (size_t)DEPTH * HID * DD * 2);
    f16*   w_fc2  = (f16*)bump(wp, (size_t)DEPTH * DD * HID * 2);
    f16*   w_head = (f16*)bump(wp, (size_t)NCLS * DD * 2);
    float* hbuf   = (float*)bump(wp, (size_t)M_TOK * DD * 4);
    f16*   xn16   = (f16*)bump(wp, (size_t)M_TOK * DD * 2);
    f16*   qkv16  = (f16*)bump(wp, (size_t)M_TOK * 3 * DD * 2);
    f16*   Qh     = (f16*)bump(wp, (size_t)BB * HEADS * SS * HDIM * 2);
    f16*   Kh     = (f16*)bump(wp, (size_t)BB * HEADS * SS * HDIM * 2);
    f16*   Vh     = (f16*)bump(wp, (size_t)BB * HEADS * SS * HDIM * 2);
    f16*   probs  = (f16*)bump(wp, (size_t)BB * HEADS * SS * SPAD * 2);
    f16*   ctx16  = (f16*)bump(wp, (size_t)M_TOK * DD * 2);
    f16*   hid16  = (f16*)bump(wp, (size_t)M_TOK * HID * 2);
    f16*   cls16  = (f16*)bump(wp, (size_t)BB * DD * 2);

    // ----- weight conversions f32 -> f16 -----
    struct { const float* s; f16* d; long long n; } convs[] = {
        { patch_w, w_pat,  (long long)DD * PD },
        { qkv_w,   w_qkv,  (long long)DEPTH * 3 * DD * DD },
        { out_w,   w_out,  (long long)DEPTH * DD * DD },
        { fc1_w,   w_fc1,  (long long)DEPTH * HID * DD },
        { fc2_w,   w_fc2,  (long long)DEPTH * DD * HID },
        { head_w,  w_head, (long long)NCLS * DD },
    };
    for (auto& c : convs)
        f32_to_f16_kernel<<<cdiv(c.n, 256), 256, 0, stream>>>(c.s, c.d, c.n);

    // ----- patch embedding -----
    im2col_kernel<<<cdiv((long long)M_PATCH * PD, 256), 256, 0, stream>>>(x, apatch);
    // tok = Apatch @ patch_w^T + patch_b, row-remapped into h[b, 1+p, :]
    gemm_wmma<EPI_F32, false><<<dim3(cdiv(DD, BN), cdiv(M_PATCH, BM), 1), 256, 0, stream>>>(
        apatch, 0, PD, w_pat, 0, PD, patch_b, nullptr,
        hbuf, 0, 0, 0, DD, 0, M_PATCH, DD, PD, PD, NPATCH, SS, 1);
    pos_cls_kernel<<<cdiv((long long)M_TOK * DD, 256), 256, 0, stream>>>(hbuf, cls_tok, pos_emb);

    const long long headStr  = (long long)SS * HDIM;   // per-(b,h) Q/K/V stride
    const long long probsStr = (long long)SS * SPAD;   // per-(b,h) scores stride

    for (int i = 0; i < DEPTH; ++i) {
        // LN1 -> xn16
        ln_kernel<<<M_TOK, 128, 0, stream>>>(hbuf, DD, ln1_g + i * DD, ln1_b + i * DD, xn16);
        // QKV = xn @ qkv_w^T + b  (f16 out)
        gemm_wmma<EPI_F16, false><<<dim3(cdiv(3 * DD, BN), cdiv(M_TOK, BM), 1), 256, 0, stream>>>(
            xn16, 0, DD, w_qkv + (size_t)i * 3 * DD * DD, 0, DD, qkv_b + i * 3 * DD, nullptr,
            qkv16, 0, 0, 0, 3 * DD, 0, M_TOK, 3 * DD, DD, DD, 0, 0, 0);
        // split into heads
        qkv_split_kernel<<<cdiv((long long)M_TOK * 3 * DD, 256), 256, 0, stream>>>(qkv16, Qh, Kh, Vh);
        // scores[bh] = Q @ K^T  (scale pre-folded), f16 out into padded rows
        gemm_wmma<EPI_F16, false><<<dim3(cdiv(SS, BN), cdiv(SS, BM), BB * HEADS), 256, 0, stream>>>(
            Qh, headStr, HDIM, Kh, headStr, HDIM, nullptr, nullptr,
            probs, probsStr, 0, 0, SPAD, 0, SS, SS, HDIM, HDIM, 0, 0, 0);
        // row softmax in-place
        softmax_kernel<<<BB * HEADS * SS, 256, 0, stream>>>(probs);
        // ctx[bh] = P @ V   (B not transposed in memory -> BT staging), f16 into [B,S,D] at col h*64
        gemm_wmma<EPI_F16, true><<<dim3(1, cdiv(SS, BM), BB * HEADS), 256, 0, stream>>>(
            probs, probsStr, SPAD, Vh, headStr, HDIM, nullptr, nullptr,
            ctx16, (long long)SS * DD, HDIM, HEADS, DD, 0, SS, HDIM, SPAD, SS, 0, 0, 0);
        // h += ctx @ out_w^T + out_b
        gemm_wmma<EPI_F32, false><<<dim3(cdiv(DD, BN), cdiv(M_TOK, BM), 1), 256, 0, stream>>>(
            ctx16, 0, DD, w_out + (size_t)i * DD * DD, 0, DD, out_b + i * DD, hbuf,
            hbuf, 0, 0, 0, DD, 0, M_TOK, DD, DD, DD, 0, 0, 0);
        // LN2 -> xn16
        ln_kernel<<<M_TOK, 128, 0, stream>>>(hbuf, DD, ln2_g + i * DD, ln2_b + i * DD, xn16);
        // hid = gelu(xn @ fc1_w^T + b)  (f16 out)
        gemm_wmma<EPI_F16_GELU, false><<<dim3(cdiv(HID, BN), cdiv(M_TOK, BM), 1), 256, 0, stream>>>(
            xn16, 0, DD, w_fc1 + (size_t)i * HID * DD, 0, DD, fc1_b + i * HID, nullptr,
            hid16, 0, 0, 0, HID, 0, M_TOK, HID, DD, DD, 0, 0, 0);
        // h += hid @ fc2_w^T + b
        gemm_wmma<EPI_F32, false><<<dim3(cdiv(DD, BN), cdiv(M_TOK, BM), 1), 256, 0, stream>>>(
            hid16, 0, HID, w_fc2 + (size_t)i * DD * HID, 0, HID, fc2_b + i * DD, hbuf,
            hbuf, 0, 0, 0, DD, 0, M_TOK, DD, HID, HID, 0, 0, 0);
    }

    // final LN on CLS rows (row stride 197*384) -> cls16, then head GEMM -> d_out
    ln_kernel<<<BB, 128, 0, stream>>>(hbuf, (long long)SS * DD, lnf_g, lnf_b, cls16);
    gemm_wmma<EPI_F32, false><<<dim3(cdiv(NCLS, BN), cdiv(BB, BM), 1), 256, 0, stream>>>(
        cls16, 0, DD, w_head, 0, DD, nullptr, nullptr,
        outp, 0, 0, 0, NCLS, 0, BB, NCLS, DD, DD, 0, 0, 0);
}
